// DisentagledAttention_27358941676234
// MI455X (gfx1250) — matrix operand order repaired and checked by reading, not compiled, not run on hardware
//
#include <hip/hip_runtime.h>
#include <hip/hip_bf16.h>
#include <math.h>

typedef __attribute__((ext_vector_type(16))) _Float16 v16h;
typedef __attribute__((ext_vector_type(8)))  float    v8f;

#define WMMA_F32_F16(a, b, c) \
  __builtin_amdgcn_wmma_f32_16x16x32_f16(false, (a), false, (b), (short)0, (c), false, false)

__device__ inline v8f vzero8() { v8f r; for (int i = 0; i < 8; ++i) r[i] = 0.0f; return r; }

// ---------------------------------------------------------------------------
// Problem constants (fixed by setup_inputs)
// ---------------------------------------------------------------------------
#define BB   4
#define SS   1024
#define DD   768
#define HH   12
#define DH   64
#define KCLIP 256
#define TT   513          // 2k+1
#define CPLD 528          // padded cp_tab row (multiple of 16)
#define RLD  520          // padded rtab row
#define D3   2304         // 3*D

// ---------------------------------------------------------------------------
// Kernel 1: TW = table[513,64] @ Wp_w[64,128]  ->  TWq[:, :64], TWk[:, 64:]
// ---------------------------------------------------------------------------
__global__ void tw_kernel(const float* __restrict__ table, const float* __restrict__ Wp,
                          float* __restrict__ TWq, float* __restrict__ TWk) {
  int idx = blockIdx.x * blockDim.x + threadIdx.x;
  if (idx >= TT * 2 * DH) return;
  int t = idx >> 7;          // / 128
  int j = idx & 127;
  float s = 0.0f;
  for (int d = 0; d < DH; ++d) s += table[t * DH + d] * Wp[d * 2 * DH + j];
  if (j < DH) TWq[t * DH + j] = s;
  else        TWk[t * DH + (j - DH)] = s;
}

// ---------------------------------------------------------------------------
// Generic batched WMMA GEMM:  C[z] = A[z] * B[z] (+bias[n]) (+bias2[z,n]),
// fp32 in, f16 compute, fp32 or f16 out. Block tile 64x64, BK=32, 4 waves
// (128 threads), each wave computes a 32x32 sub-tile as 2x2 WMMA fragments.
//   A element (m,k) = A[z/nH*sAb + z%nH*sAh + m*lda + k]
//   B element (k,n) = transB ? B[z*sBz + n*ldb + k] : B[z*sBz + k*ldb + n]
//   C element (m,n) = C[z*sCz + m*ldc + n]
//   bias2 element   = bias2[z*sB2z + n]          (per-batch additive row)
// ---------------------------------------------------------------------------
template <bool HOUT>
__global__ void gemm_wmma(const float* __restrict__ A, int lda, long sAb, long sAh, int nH,
                          const float* __restrict__ Bm, int ldb, int transB, long sBz,
                          void* __restrict__ Cout, int ldc, long sCz,
                          const float* __restrict__ bias,
                          const float* __restrict__ bias2, long sB2z,
                          int M, int N, int K) {
  __shared__ _Float16 As[64][32];    // [m][k]
  __shared__ _Float16 BsT[64][32];   // [n][k]  (B staged transposed)

  const int z = blockIdx.z;
  const float* Ab = A + (long)(z / nH) * sAb + (long)(z % nH) * sAh;
  const float* Bb = Bm + (long)z * sBz;
  const int m0 = blockIdx.y * 64;
  const int n0 = blockIdx.x * 64;
  const int tid  = threadIdx.x;      // 0..127
  const int wid  = tid >> 5;         // 0..3
  const int lane = tid & 31;
  const int wm = (wid >> 1) * 32;
  const int wn = (wid & 1) * 32;
  const int lrow = lane & 15;
  const int lhi  = lane >> 4;        // 0 or 1

  v8f acc[2][2];
  for (int i = 0; i < 2; ++i)
    for (int j = 0; j < 2; ++j) acc[i][j] = vzero8();

  const int ksteps = (K + 31) / 32;
  for (int kb = 0; kb < ksteps; ++kb) {
    const int k0 = kb * 32;
    // software prefetch of the next K-tile (global_prefetch_b8)
    if (kb + 1 < ksteps) {
      const int k1 = k0 + 32;
      if (tid < 64) {
        int gm = m0 + tid;
        if (gm < M && k1 < K) __builtin_prefetch(Ab + (long)gm * lda + k1, 0, 0);
      } else {
        int gn = n0 + (tid - 64);
        if (gn < N && k1 < K) {
          const float* p = transB ? (Bb + (long)gn * ldb + k1)
                                  : (Bb + (long)k1 * ldb + gn);
          __builtin_prefetch(p, 0, 0);
        }
      }
    }
    // stage A tile (64x32) as f16
    for (int idx = tid; idx < 64 * 32; idx += 128) {
      int r = idx >> 5, c = idx & 31;
      int gm = m0 + r, gk = k0 + c;
      float v = (gm < M && gk < K) ? Ab[(long)gm * lda + gk] : 0.0f;
      As[r][c] = (_Float16)v;
    }
    // stage B tile transposed (64 n x 32 k) as f16
    for (int idx = tid; idx < 64 * 32; idx += 128) {
      int n = idx >> 5, c = idx & 31;
      int gn = n0 + n, gk = k0 + c;
      float v = 0.0f;
      if (gn < N && gk < K)
        v = transB ? Bb[(long)gn * ldb + gk] : Bb[(long)gk * ldb + gn];
      BsT[n][c] = (_Float16)v;
    }
    __syncthreads();

    // build A fragments (rows wm, wm+16), documented 16-bit A layout
    v16h afrag[2];
    const int kb8 = lhi * 8;
    for (int i = 0; i < 2; ++i) {
      int row = wm + 16 * i + lrow;
      for (int e = 0; e < 8; ++e) afrag[i][e]     = As[row][kb8 + e];
      for (int e = 0; e < 8; ++e) afrag[i][8 + e] = As[row][16 + kb8 + e];
    }
    // build B fragments (cols wn, wn+16)
    v16h bfrag[2];
    for (int j = 0; j < 2; ++j) {
      int n = wn + 16 * j + lrow;
      for (int e = 0; e < 8; ++e) bfrag[j][e]     = BsT[n][kb8 + e];
      for (int e = 0; e < 8; ++e) bfrag[j][8 + e] = BsT[n][16 + kb8 + e];
    }
    for (int i = 0; i < 2; ++i)
      for (int j = 0; j < 2; ++j)
        acc[i][j] = WMMA_F32_F16(afrag[i], bfrag[j], acc[i][j]);
    __syncthreads();
  }

  // store: D layout -> row = base + v + 8*lhi, col = base + lane%16
  for (int i = 0; i < 2; ++i)
    for (int j = 0; j < 2; ++j)
      for (int v = 0; v < 8; ++v) {
        int row = m0 + wm + 16 * i + v + 8 * lhi;
        int col = n0 + wn + 16 * j + lrow;
        if (row < M && col < N) {
          float val = acc[i][j][v];
          if (bias)  val += bias[col];
          if (bias2) val += bias2[(long)z * sB2z + col];
          long off = (long)z * sCz + (long)row * ldc + col;
          if (HOUT) ((_Float16*)Cout)[off] = (_Float16)val;
          else      ((float*)Cout)[off]    = val;
        }
      }
}

// ---------------------------------------------------------------------------
// Kernel 3: kc_sum[b,h,d] = sum_s kc[b,h,s,d]
// ---------------------------------------------------------------------------
__global__ void kcsum_kernel(const float* __restrict__ qkv, float* __restrict__ kcsum) {
  int bh = blockIdx.x;
  int b = bh / HH, h = bh % HH;
  int d = threadIdx.x;   // 64 threads
  float s = 0.0f;
  for (int j = 0; j < SS; ++j)
    s += qkv[((long)(b * SS + j)) * D3 + DD + h * DH + d];
  kcsum[bh * DH + d] = s;
}

// ---------------------------------------------------------------------------
// Kernel 4: rtab[bh,t] = TWq[t,:] . kcsum[bh,:]
// ---------------------------------------------------------------------------
__global__ void rtab_kernel(const float* __restrict__ TWq, const float* __restrict__ kcsum,
                            float* __restrict__ rtab) {
  __shared__ float ks[DH];
  int bh = blockIdx.x;
  if (threadIdx.x < DH) ks[threadIdx.x] = kcsum[bh * DH + threadIdx.x];
  __syncthreads();
  for (int t = threadIdx.x; t < TT; t += blockDim.x) {
    float s = 0.0f;
    for (int d = 0; d < DH; ++d) s += TWq[t * DH + d] * ks[d];
    rtab[bh * RLD + t] = s;
  }
}

// ---------------------------------------------------------------------------
// Kernel 6: flash attention. One wave (32 threads) per (b,h, 16-row Q tile).
// Online softmax over s (1024) in blocks of 32 columns.
// cp_tab already contains c_p + c_r (rtab folded in as GEMM bias2).
// ---------------------------------------------------------------------------
__global__ void __launch_bounds__(32)
flash_attn(const float* __restrict__ qkv, const _Float16* __restrict__ cp_tab,
           const int* __restrict__ amask, float* __restrict__ attn_out) {
  __shared__ _Float16 Kt[32][DH];   // K tile rows (also reused for Q staging)
  __shared__ _Float16 Vt[DH][32];   // V tile transposed [d][s]
  __shared__ _Float16 Pt[16][32];   // softmax probabilities (A-matrix staging)

  const int bh = blockIdx.y;
  const int b = bh / HH, h = bh % HH;
  const int i0 = blockIdx.x * 16;
  const int lane = threadIdx.x;
  const int lrow = lane & 15;
  const int lhi  = lane >> 4;

  // ---- stage Q tile (16x64) as f16 (vectorized) and build A fragments ----
  for (int idx = lane; idx < 16 * 16; idx += 32) {      // 16 rows x 16 float4
    int r = idx >> 4, c4 = (idx & 15) * 4;
    const float4 f = *(const float4*)(qkv + ((long)(b * SS + i0 + r)) * D3 + h * DH + c4);
    Kt[r][c4 + 0] = (_Float16)f.x; Kt[r][c4 + 1] = (_Float16)f.y;
    Kt[r][c4 + 2] = (_Float16)f.z; Kt[r][c4 + 3] = (_Float16)f.w;
  }
  __syncthreads();
  v16h a_q[2];
  {
    const int kb8 = lhi * 8;
    for (int i = 0; i < 2; ++i) {
      int dbase = 32 * i;
      for (int e = 0; e < 8; ++e) a_q[i][e]     = Kt[lrow][dbase + kb8 + e];
      for (int e = 0; e < 8; ++e) a_q[i][8 + e] = Kt[lrow][dbase + 16 + kb8 + e];
    }
  }
  __syncthreads();

  float mrow[8], lsum[8];
  for (int v = 0; v < 8; ++v) { mrow[v] = -1.0e30f; lsum[v] = 0.0f; }
  v8f accO[4];
  for (int nt = 0; nt < 4; ++nt) accO[nt] = vzero8();

  const long cp_base = (long)bh * SS * CPLD;

  for (int s0 = 0; s0 < SS; s0 += 32) {
    // ---- stage K (32x64) and V (transposed 64x32), float4 loads ----
    for (int idx = lane; idx < 32 * 16; idx += 32) {    // 32 rows x 16 float4
      int r = idx >> 4, c4 = (idx & 15) * 4;
      const float* row = qkv + (long)(b * SS + s0 + r) * D3 + h * DH;
      const float4 fk = *(const float4*)(row + DD + c4);
      const float4 fv = *(const float4*)(row + 2 * DD + c4);
      Kt[r][c4 + 0] = (_Float16)fk.x; Kt[r][c4 + 1] = (_Float16)fk.y;
      Kt[r][c4 + 2] = (_Float16)fk.z; Kt[r][c4 + 3] = (_Float16)fk.w;
      Vt[c4 + 0][r] = (_Float16)fv.x; Vt[c4 + 1][r] = (_Float16)fv.y;
      Vt[c4 + 2][r] = (_Float16)fv.z; Vt[c4 + 3][r] = (_Float16)fv.w;
    }
    // ---- prefetch next s-block's K/V rows (global_prefetch_b8) ----
    if (s0 + 32 < SS) {
      const float* nrow = qkv + (long)(b * SS + s0 + 32 + lane) * D3 + h * DH;
      __builtin_prefetch(nrow + DD, 0, 0);
      __builtin_prefetch(nrow + DD + 32, 0, 0);
      __builtin_prefetch(nrow + 2 * DD, 0, 0);
      __builtin_prefetch(nrow + 2 * DD + 32, 0, 0);
    }
    __syncthreads();

    const int mv0 = amask[b * SS + s0 + lrow];
    const int mv1 = amask[b * SS + s0 + 16 + lrow];

    // ---- score S = Q Kt^T (two 16-col D tiles) + (c_p + c_r) gather ----
    float sv[2][8];
    for (int j = 0; j < 2; ++j) {
      v16h bk[2];
      const int kb8 = lhi * 8;
      for (int i = 0; i < 2; ++i)
        for (int e = 0; e < 8; ++e) {
          bk[i][e]     = Kt[16 * j + lrow][32 * i + kb8 + e];
          bk[i][8 + e] = Kt[16 * j + lrow][32 * i + 16 + kb8 + e];
        }
      v8f sacc = vzero8();
      sacc = WMMA_F32_F16(a_q[0], bk[0], sacc);
      sacc = WMMA_F32_F16(a_q[1], bk[1], sacc);
      const int scol = s0 + 16 * j + lrow;
      for (int v = 0; v < 8; ++v) {
        int irow = i0 + v + 8 * lhi;
        int t = irow - scol;
        t = (t > KCLIP ? KCLIP : (t < -KCLIP ? -KCLIP : t)) + KCLIP;
        float val = sacc[v] + (float)cp_tab[cp_base + (long)irow * CPLD + t];
        val *= 0.125f;   // 1/sqrt(64)
        if ((j ? mv1 : mv0) == 0) val = -1.0e9f;
        sv[j][v] = val;
      }
    }

    // ---- online softmax (reductions within each 16-lane half = one D-tile) ----
    for (int v = 0; v < 8; ++v) {
      float mx = fmaxf(sv[0][v], sv[1][v]);
      for (int m = 1; m < 16; m <<= 1) mx = fmaxf(mx, __shfl_xor(mx, m, 32));
      float mnew  = fmaxf(mrow[v], mx);
      float alpha = __expf(mrow[v] - mnew);
      float p0 = __expf(sv[0][v] - mnew);
      float p1 = __expf(sv[1][v] - mnew);
      float rs = p0 + p1;
      for (int m = 1; m < 16; m <<= 1) rs += __shfl_xor(rs, m, 32);
      lsum[v] = lsum[v] * alpha + rs;
      mrow[v] = mnew;
      for (int nt = 0; nt < 4; ++nt) accO[nt][v] = accO[nt][v] * alpha;
      int prow = v + 8 * lhi;
      Pt[prow][lrow]      = (_Float16)p0;
      Pt[prow][16 + lrow] = (_Float16)p1;
    }
    __syncthreads();

    // ---- O += P (16x32) x V (32x64), 4 WMMAs over d-head tiles ----
    v16h a_p;
    {
      const int kb8 = lhi * 8;
      for (int e = 0; e < 8; ++e) a_p[e]     = Pt[lrow][kb8 + e];
      for (int e = 0; e < 8; ++e) a_p[8 + e] = Pt[lrow][16 + kb8 + e];
    }
    const int kb8 = lhi * 8;
    for (int nt = 0; nt < 4; ++nt) {
      v16h bv;
      for (int e = 0; e < 8; ++e) bv[e]     = Vt[nt * 16 + lrow][kb8 + e];
      for (int e = 0; e < 8; ++e) bv[8 + e] = Vt[nt * 16 + lrow][16 + kb8 + e];
      accO[nt] = WMMA_F32_F16(a_p, bv, accO[nt]);
    }
    __syncthreads();
  }

  // ---- epilogue: normalize and store in [B,S,H*dh] layout ----
  for (int nt = 0; nt < 4; ++nt)
    for (int v = 0; v < 8; ++v) {
      int row = i0 + v + 8 * lhi;
      int d   = h * DH + nt * 16 + lrow;
      attn_out[((long)(b * SS + row)) * DD + d] = accO[nt][v] / lsum[v];
    }
}

// ---------------------------------------------------------------------------
// Host launcher
// ---------------------------------------------------------------------------
extern "C" void kernel_launch(void* const* d_in, const int* in_sizes, int n_in,
                              void* d_out, int out_size, void* d_ws, size_t ws_size,
                              hipStream_t stream) {
  (void)in_sizes; (void)n_in; (void)out_size; (void)ws_size;
  const float* x       = (const float*)d_in[0];
  const int*   amask   = (const int*)  d_in[1];
  const float* Wc_w    = (const float*)d_in[2];
  const float* Wc_b    = (const float*)d_in[3];
  const float* Wp_w    = (const float*)d_in[4];
  const float* table   = (const float*)d_in[5];
  const float* cproj_w = (const float*)d_in[6];
  const float* cproj_b = (const float*)d_in[7];

  // workspace carve-up (floats, then halves)
  float* qkv   = (float*)d_ws;                    // 4*1024*2304
  float* TWq   = qkv   + (long)BB * SS * D3;      // 528*64 (513 used)
  float* TWk   = TWq   + CPLD * DH;
  float* kcsum = TWk   + CPLD * DH;               // 48*64
  float* rtab  = kcsum + BB * HH * DH;            // 48*520
  float* attnO = rtab  + BB * HH * RLD;           // 4*1024*768
  _Float16* cp = (_Float16*)(attnO + (long)BB * SS * DD); // 48*1024*528 halves

  // 1) TW = table @ Wp_w
  tw_kernel<<<(TT * 2 * DH + 255) / 256, 256, 0, stream>>>(table, Wp_w, TWq, TWk);

  // 2) qkv = x @ Wc_w + Wc_b   (M=4096, N=2304, K=768)
  gemm_wmma<false><<<dim3(D3 / 64, (BB * SS) / 64, 1), 128, 0, stream>>>(
      x, DD, 0L, 0L, 1, Wc_w, D3, 0, 0L, (void*)qkv, D3, 0L, Wc_b,
      nullptr, 0L, BB * SS, D3, DD);

  // 3) kc_sum
  kcsum_kernel<<<BB * HH, DH, 0, stream>>>(qkv, kcsum);

  // 4) rtab[bh,t] = TWq[t] . kcsum[bh]
  rtab_kernel<<<BB * HH, 128, 0, stream>>>(TWq, kcsum, rtab);

  // 5) cp_tab[bh,i,t] = qc[bh,i,:] . TWk[t,:] + rtab[bh,t]
  //    (batched, N=513, K=64, f16 out, rtab folded in as per-batch bias2)
  gemm_wmma<true><<<dim3((TT + 63) / 64, SS / 64, BB * HH), 128, 0, stream>>>(
      qkv, D3, (long)SS * D3, (long)DH, HH,
      TWk, DH, 1, 0L, (void*)cp, CPLD, (long)SS * CPLD, nullptr,
      rtab, (long)RLD, SS, TT, DH);

  // 6) flash attention
  flash_attn<<<dim3(SS / 16, BB * HH), 32, 0, stream>>>(qkv, cp, amask, attnO);

  // 7) out = attnO @ cproj_w + cproj_b   (M=4096, N=768, K=768)
  gemm_wmma<false><<<dim3(DD / 64, (BB * SS) / 64, 1), 128, 0, stream>>>(
      attnO, DD, 0L, 0L, 1, cproj_w, DD, 0, 0L, d_out, DD, 0L, cproj_b,
      nullptr, 0L, BB * SS, DD, DD);
}
